// GridificationLayer_28930899706114
// MI455X (gfx1250) — matrix operand order repaired and checked by reading, not compiled, not run on hardware
//
#include <hip/hip_runtime.h>
#include <hip/hip_bf16.h>
#include <math.h>

typedef _Float16 half_t;
typedef __attribute__((ext_vector_type(16))) _Float16 v16h;
typedef __attribute__((ext_vector_type(8)))  _Float16 v8h;
typedef __attribute__((ext_vector_type(8)))  float    v8f;

#define NN 100000
#define GG 32768
#define EE 1000000
#define PW 72   // padded LDS row pitch (halves) for 64-wide matrices

// ---------------- WMMA fragment helpers (CDNA5 wave32, 16x16x32 f16) --------
// 16-bit A (16x32, MxK) per ISA 7.12.2:
//   lane<16:  row=lane,     elems 0..7 = K 0..7,   8..15 = K 16..23
//   lane>=16: row=lane-16,  elems 0..7 = K 8..15,  8..15 = K 24..31
__device__ __forceinline__ v16h frag_a_ptr(const half_t* rowbase_k0) {
  const int lane = threadIdx.x & 31;
  const half_t* q = rowbase_k0 + ((lane >> 4) << 3);
  v8h lo = *(const v8h*)q;
  v8h hi = *(const v8h*)(q + 16);
  return __builtin_shufflevector(lo, hi, 0,1,2,3,4,5,6,7,8,9,10,11,12,13,14,15);
}
__device__ __forceinline__ v16h frag_a(const half_t* base_k0, int ld) {
  const int lane = threadIdx.x & 31;
  return frag_a_ptr(base_k0 + (lane & 15) * ld);
}
// 16-bit B (32x16, KxN): col = lane&15; elems 0..7 = K 8h+0..7, 8..15 = K 16+8h+0..7.
__device__ __forceinline__ v16h frag_b(const half_t* base /*[k0][n0]*/, int ld) {
  const int lane = threadIdx.x & 31;
  const half_t* q = base + (lane & 15) + ((lane >> 4) << 3) * ld;
  v16h r;
#pragma unroll
  for (int i = 0; i < 8; ++i) {
    r[i] = q[i * ld];
    r[8 + i] = q[(16 + i) * ld];
  }
  return r;
}
__device__ __forceinline__ v8f wmma32(v16h a, v16h b, v8f c) {
  return __builtin_amdgcn_wmma_f32_16x16x32_f16(false, a, false, b, (short)0, c,
                                                false, false);
}
// Fast SiLU: x * sigmoid(x) with hardware v_rcp_f32 instead of IEEE divide
// (the precise div lowers to a ~10-op div_scale/div_fma sequence; rcp is one
// TRANS op and far exceeds the f16 storage precision downstream).
__device__ __forceinline__ float silu(float x) {
  return x * __builtin_amdgcn_rcpf(1.0f + __expf(-x));
}
// Wave-local LDS producer->consumer sync: per-wave DS ops are in order; after
// DScnt drains, this wave's LDS writes are visible to all of its lanes.
__device__ __forceinline__ void lds_wave_sync() {
  __asm__ volatile("s_wait_dscnt 0x0" ::: "memory");
}
// k-outer / nt-inner GEMM: per k-step, load the 4 B fragments for the whole
// N row, one DS wait, then 4 back-to-back WMMAs sharing one A fragment.
template <int KT>
__device__ __forceinline__ void gemm_acc(v8f acc[4], const v16h* a,
                                         const half_t* W) {
#pragma unroll
  for (int k = 0; k < KT; ++k) {
    const half_t* Wk = W + k * 32 * PW;
    v16h b0 = frag_b(Wk + 0, PW);
    v16h b1 = frag_b(Wk + 16, PW);
    v16h b2 = frag_b(Wk + 32, PW);
    v16h b3 = frag_b(Wk + 48, PW);
    acc[0] = wmma32(a[k], b0, acc[0]);
    acc[1] = wmma32(a[k], b1, acc[1]);
    acc[2] = wmma32(a[k], b2, acc[2]);
    acc[3] = wmma32(a[k], b3, acc[3]);
  }
}
// Cooperative copy of a rows x 64 f16 matrix (global, pitch 64) into LDS with
// padded pitch PW=72 (bank-conflict-free lane-strided B-fragment reads).
__device__ __forceinline__ void stage_w(const half_t* __restrict__ g,
                                        half_t* s, int rows, int tid,
                                        int nthreads) {
  const int total = rows * 8;  // 8 x v8h chunks per row
  for (int i = tid; i < total; i += nthreads) {
    int r = i >> 3, c = (i & 7) << 3;
    *(v8h*)(s + r * PW + c) = *(const v8h*)(g + r * 64 + c);
  }
}

// ---------------- prep kernels ----------------------------------------------
__global__ void k_cvt(const float* __restrict__ in, half_t* __restrict__ out,
                      int n) {
  int i = blockIdx.x * blockDim.x + threadIdx.x;
  if (i < n) out[i] = (half_t)in[i];
}
// edge_w1 padded [6][64] -> [32][64]; row 6 carries edge_b1 (bias folded into
// the K padding: edge_attr element 6 is the constant 1.0).
__global__ void k_prep_we1(const float* __restrict__ w,
                           const float* __restrict__ b,
                           half_t* __restrict__ out) {
  int i = blockIdx.x * blockDim.x + threadIdx.x;
  if (i < 32 * 64) {
    int k = i >> 6, n = i & 63;
    float v = (k < 6) ? w[i] : (k == 6 ? b[n] : 0.0f);
    out[i] = (half_t)v;
  }
}
__global__ void k_zero(float* __restrict__ p, int n) {
  int i = blockIdx.x * blockDim.x + threadIdx.x;
  if (i < n) p[i] = 0.0f;
}

// ---------------- node MLP: h_node = L2(silu(L1(x))) as f16 [N,64] ----------
__global__ void __launch_bounds__(256) k_node_mlp(
    const half_t* __restrict__ xh, const half_t* __restrict__ w1,
    const float* __restrict__ b1, const half_t* __restrict__ w2,
    const float* __restrict__ b2, half_t* __restrict__ hnode) {
  __shared__ __align__(16) half_t sw1[64 * PW];
  __shared__ __align__(16) half_t sw2[64 * PW];
  __shared__ float sb[2 * 64];
  __shared__ __align__(16) half_t sh[8][16 * PW];
  const int tid = threadIdx.x;
  stage_w(w1, sw1, 64, tid, 256);
  stage_w(w2, sw2, 64, tid, 256);
  if (tid < 64) { sb[tid] = b1[tid]; sb[64 + tid] = b2[tid]; }
  __syncthreads();

  const int lane = tid & 31, wv = tid >> 5;
  const int col = lane & 15, rbase = (lane >> 4) << 3;
  float bv1[4], bv2[4];
#pragma unroll
  for (int nt = 0; nt < 4; ++nt) {
    bv1[nt] = sb[nt * 16 + col];
    bv2[nt] = sb[64 + nt * 16 + col];
  }
  const int NT = NN / 16;  // 6250
  for (int tile = blockIdx.x * 8 + wv; tile < NT; tile += gridDim.x * 8) {
    const half_t* ab = xh + (size_t)tile * 16 * 64;
    v16h a[2] = {frag_a(ab, 64), frag_a(ab + 32, 64)};
    v8f acc[4] = {{0,0,0,0,0,0,0,0}, {0,0,0,0,0,0,0,0},
                  {0,0,0,0,0,0,0,0}, {0,0,0,0,0,0,0,0}};
    gemm_acc<2>(acc, a, sw1);
#pragma unroll
    for (int nt = 0; nt < 4; ++nt)
#pragma unroll
      for (int r = 0; r < 8; ++r)
        sh[wv][(rbase + r) * PW + nt * 16 + col] =
            (half_t)silu(acc[nt][r] + bv1[nt]);
    lds_wave_sync();
    v16h h[2] = {frag_a(&sh[wv][0], PW), frag_a(&sh[wv][0] + 32, PW)};
    v8f a2[4] = {{0,0,0,0,0,0,0,0}, {0,0,0,0,0,0,0,0},
                 {0,0,0,0,0,0,0,0}, {0,0,0,0,0,0,0,0}};
    gemm_acc<2>(a2, h, sw2);
#pragma unroll
    for (int nt = 0; nt < 4; ++nt)
#pragma unroll
      for (int r = 0; r < 8; ++r)
        hnode[(size_t)(tile * 16 + rbase + r) * 64 + nt * 16 + col] =
            (half_t)(a2[nt][r] + bv2[nt]);
    lds_wave_sync();
  }
}

// ---------------- fused edge stage (persistent, 4 waves/block) --------------
__global__ void __launch_bounds__(128) k_edge(
    const float* __restrict__ node_pos, const float* __restrict__ grid_pos,
    const float* __restrict__ orient, const int* __restrict__ src,
    const int* __restrict__ tgt, const half_t* __restrict__ hnode,
    const half_t* __restrict__ we1, const half_t* __restrict__ we2,
    const float* __restrict__ eb2, const half_t* __restrict__ wm1,
    const float* __restrict__ mb1, const half_t* __restrict__ wm2,
    const float* __restrict__ mb2, float* __restrict__ sums,
    float* __restrict__ cnts) {
  __shared__ __align__(16) half_t swe1[32 * PW];
  __shared__ __align__(16) half_t swe2[64 * PW];
  __shared__ __align__(16) half_t swm1[128 * PW];
  __shared__ __align__(16) half_t swm2[64 * PW];
  __shared__ float sb[3 * 64];
  __shared__ __align__(16) half_t ef[4][16 * PW];  // hidden acts (reused)
  __shared__ __align__(16) half_t em[4][16 * PW];  // e_feat; ea overlays here
  const int tid = threadIdx.x;
  stage_w(we1, swe1, 32, tid, 128);
  stage_w(we2, swe2, 64, tid, 128);
  stage_w(wm1, swm1, 128, tid, 128);
  stage_w(wm2, swm2, 64, tid, 128);
  if (tid < 64) {
    sb[tid] = eb2[tid];
    sb[64 + tid] = mb1[tid];
    sb[128 + tid] = mb2[tid];
  }
  __syncthreads();

  const int lane = tid & 31, wv = tid >> 5;
  const int col = lane & 15, rbase = (lane >> 4) << 3;
  float be2[4], bm1[4], bm2[4];
#pragma unroll
  for (int nt = 0; nt < 4; ++nt) {
    be2[nt] = sb[nt * 16 + col];
    bm1[nt] = sb[64 + nt * 16 + col];
    bm2[nt] = sb[128 + nt * 16 + col];
  }
  // edge_attr (16x32, pitch 40) overlays the e_feat slab: ea is dead before
  // em is first written; per-wave in-order DS keeps iterations safe.
  half_t* ea_s = &em[wv][0];
  const int NT = EE / 16;  // 62500
  for (int tile = blockIdx.x * 4 + wv; tile < NT; tile += gridDim.x * 4) {
    const int e0 = tile * 16;
    // --- per-edge geometry (one edge per low lane) ---
    if (lane < 16) {
      int e = e0 + lane;
      int s = src[e];
      int t = tgt[e];
      int tc = t < 0 ? 0 : (t >= GG ? GG - 1 : t);
      float ps0 = node_pos[s * 3 + 0], ps1 = node_pos[s * 3 + 1],
            ps2 = node_pos[s * 3 + 2];
      float r0 = grid_pos[tc * 3 + 0] - ps0;
      float r1 = grid_pos[tc * 3 + 1] - ps1;
      float r2 = grid_pos[tc * 3 + 2] - ps2;
      const float* R = orient + (size_t)s * 9;
      float t0 = r0 * R[0] + r1 * R[3] + r2 * R[6];
      float t1 = r0 * R[1] + r1 * R[4] + r2 * R[7];
      float t2 = r0 * R[2] + r1 * R[5] + r2 * R[8];
      // element 6 = 1.0 multiplies the bias row folded into we1
      v8h row = {(half_t)ps0, (half_t)ps1, (half_t)ps2, (half_t)t0,
                 (half_t)t1,  (half_t)t2,  (half_t)1.0f, (half_t)0.0f};
      v8h z = {};
      *(v8h*)(ea_s + lane * 40 + 0) = row;
      *(v8h*)(ea_s + lane * 40 + 8) = z;
      *(v8h*)(ea_s + lane * 40 + 16) = z;
      *(v8h*)(ea_s + lane * 40 + 24) = z;
    }
    // --- fragment-gather h_node[source] (L2-resident f16 table) ---
    int rsrc = src[e0 + col];
    const half_t* hb = hnode + (size_t)rsrc * 64;
    v16h ha0 = frag_a_ptr(hb);       // msg-in K 0..31
    v16h ha1 = frag_a_ptr(hb + 32);  // msg-in K 32..63
    lds_wave_sync();

    // --- edge MLP L1 (bias folded into weights): silu -> ef ---
    {
      v16h a[1] = {frag_a(ea_s, 40)};
      v8f acc[4] = {{0,0,0,0,0,0,0,0}, {0,0,0,0,0,0,0,0},
                    {0,0,0,0,0,0,0,0}, {0,0,0,0,0,0,0,0}};
      gemm_acc<1>(acc, a, swe1);
#pragma unroll
      for (int nt = 0; nt < 4; ++nt)
#pragma unroll
        for (int r = 0; r < 8; ++r)
          ef[wv][(rbase + r) * PW + nt * 16 + col] = (half_t)silu(acc[nt][r]);
    }
    lds_wave_sync();

    // --- edge MLP L2: e_feat -> em (overwrites dead ea region) ---
    {
      v16h a[2] = {frag_a(&ef[wv][0], PW), frag_a(&ef[wv][0] + 32, PW)};
      v8f acc[4] = {{0,0,0,0,0,0,0,0}, {0,0,0,0,0,0,0,0},
                    {0,0,0,0,0,0,0,0}, {0,0,0,0,0,0,0,0}};
      gemm_acc<2>(acc, a, swe2);
#pragma unroll
      for (int nt = 0; nt < 4; ++nt)
#pragma unroll
        for (int r = 0; r < 8; ++r)
          em[wv][(rbase + r) * PW + nt * 16 + col] =
              (half_t)(acc[nt][r] + be2[nt]);
    }
    lds_wave_sync();

    // --- msg MLP L1: K=128 = [h_node[src] | e_feat], silu -> ef ---
    {
      v16h a[4];
      a[0] = ha0;
      a[1] = ha1;
      a[2] = frag_a(&em[wv][0], PW);
      a[3] = frag_a(&em[wv][0] + 32, PW);
      v8f acc[4] = {{0,0,0,0,0,0,0,0}, {0,0,0,0,0,0,0,0},
                    {0,0,0,0,0,0,0,0}, {0,0,0,0,0,0,0,0}};
      gemm_acc<4>(acc, a, swm1);
#pragma unroll
      for (int nt = 0; nt < 4; ++nt)
#pragma unroll
        for (int r = 0; r < 8; ++r)
          ef[wv][(rbase + r) * PW + nt * 16 + col] =
              (half_t)silu(acc[nt][r] + bm1[nt]);
    }
    lds_wave_sync();

    // --- msg MLP L2 + scatter-add (f32 atomics resolve in L2) ---
    {
      v16h a[2] = {frag_a(&ef[wv][0], PW), frag_a(&ef[wv][0] + 32, PW)};
      v8f acc[4] = {{0,0,0,0,0,0,0,0}, {0,0,0,0,0,0,0,0},
                    {0,0,0,0,0,0,0,0}, {0,0,0,0,0,0,0,0}};
      gemm_acc<2>(acc, a, swm2);
      int trow[8];
#pragma unroll
      for (int r = 0; r < 8; ++r) trow[r] = tgt[e0 + rbase + r];
#pragma unroll
      for (int nt = 0; nt < 4; ++nt)
#pragma unroll
        for (int r = 0; r < 8; ++r) {
          int t = trow[r];
          if ((unsigned)t < (unsigned)GG)
            atomicAdd(&sums[(size_t)t * 64 + nt * 16 + col],
                      acc[nt][r] + bm2[nt]);
        }
      if (lane < 16) {
        int t = tgt[e0 + lane];
        if ((unsigned)t < (unsigned)GG) atomicAdd(&cnts[t], 1.0f);
      }
    }
    lds_wave_sync();
  }
}

// ---------------- finalize: mean + update MLP -> out [G,64] f32 -------------
__global__ void __launch_bounds__(256) k_finalize(
    const float* __restrict__ sums, const float* __restrict__ cnts,
    const half_t* __restrict__ w1, const float* __restrict__ b1,
    const half_t* __restrict__ w2, const float* __restrict__ b2,
    float* __restrict__ out) {
  __shared__ __align__(16) half_t sw1[64 * PW];
  __shared__ __align__(16) half_t sw2[64 * PW];
  __shared__ float sb[2 * 64];
  __shared__ __align__(16) half_t gf[8][16 * PW];
  __shared__ __align__(16) half_t hh[8][16 * PW];
  __shared__ float inv[8][16];
  const int tid = threadIdx.x;
  stage_w(w1, sw1, 64, tid, 256);
  stage_w(w2, sw2, 64, tid, 256);
  if (tid < 64) { sb[tid] = b1[tid]; sb[64 + tid] = b2[tid]; }
  __syncthreads();

  const int lane = tid & 31, wv = tid >> 5;
  const int col = lane & 15, rbase = (lane >> 4) << 3;
  float bv1[4], bv2[4];
#pragma unroll
  for (int nt = 0; nt < 4; ++nt) {
    bv1[nt] = sb[nt * 16 + col];
    bv2[nt] = sb[64 + nt * 16 + col];
  }
  const int NT = GG / 16;  // 2048
  for (int tile = blockIdx.x * 8 + wv; tile < NT; tile += gridDim.x * 8) {
    const int r0 = tile * 16;
    if (lane < 16) inv[wv][lane] = __builtin_amdgcn_rcpf(
        fmaxf(cnts[r0 + lane], 1.0f));
    lds_wave_sync();
    for (int i = lane; i < 1024; i += 32) {
      int rr = i >> 6, cc = i & 63;
      gf[wv][rr * PW + cc] =
          (half_t)(sums[(size_t)(r0 + rr) * 64 + cc] * inv[wv][rr]);
    }
    lds_wave_sync();
    v16h a[2] = {frag_a(&gf[wv][0], PW), frag_a(&gf[wv][0] + 32, PW)};
    v8f acc[4] = {{0,0,0,0,0,0,0,0}, {0,0,0,0,0,0,0,0},
                  {0,0,0,0,0,0,0,0}, {0,0,0,0,0,0,0,0}};
    gemm_acc<2>(acc, a, sw1);
#pragma unroll
    for (int nt = 0; nt < 4; ++nt)
#pragma unroll
      for (int r = 0; r < 8; ++r)
        hh[wv][(rbase + r) * PW + nt * 16 + col] =
            (half_t)silu(acc[nt][r] + bv1[nt]);
    lds_wave_sync();
    v16h h[2] = {frag_a(&hh[wv][0], PW), frag_a(&hh[wv][0] + 32, PW)};
    v8f a2[4] = {{0,0,0,0,0,0,0,0}, {0,0,0,0,0,0,0,0},
                 {0,0,0,0,0,0,0,0}, {0,0,0,0,0,0,0,0}};
    gemm_acc<2>(a2, h, sw2);
#pragma unroll
    for (int nt = 0; nt < 4; ++nt)
#pragma unroll
      for (int r = 0; r < 8; ++r)
        out[(size_t)(r0 + rbase + r) * 64 + nt * 16 + col] =
            a2[nt][r] + bv2[nt];
    lds_wave_sync();
  }
}

// ---------------- host launcher ---------------------------------------------
extern "C" void kernel_launch(void* const* d_in, const int* in_sizes, int n_in,
                              void* d_out, int out_size, void* d_ws,
                              size_t ws_size, hipStream_t stream) {
  (void)in_sizes; (void)n_in; (void)out_size; (void)ws_size;
  const float* node_features = (const float*)d_in[0];
  const float* node_pos = (const float*)d_in[1];
  const float* grid_pos = (const float*)d_in[2];
  const float* orient = (const float*)d_in[3];
  const int* eidx = (const int*)d_in[4];
  const float* nw1 = (const float*)d_in[5];
  const float* nb1 = (const float*)d_in[6];
  const float* nw2 = (const float*)d_in[7];
  const float* nb2 = (const float*)d_in[8];
  const float* ew1 = (const float*)d_in[9];
  const float* eb1 = (const float*)d_in[10];
  const float* ew2 = (const float*)d_in[11];
  const float* eb2 = (const float*)d_in[12];
  const float* mw1 = (const float*)d_in[13];
  const float* mb1 = (const float*)d_in[14];
  const float* mw2 = (const float*)d_in[15];
  const float* mb2 = (const float*)d_in[16];
  const float* uw1 = (const float*)d_in[17];
  const float* ub1 = (const float*)d_in[18];
  const float* uw2 = (const float*)d_in[19];
  const float* ub2 = (const float*)d_in[20];

  char* ws = (char*)d_ws;
  size_t off = 0;
  auto take = [&](size_t bytes) -> char* {
    off = (off + 255) & ~(size_t)255;
    char* p = ws + off;
    off += bytes;
    return p;
  };
  half_t* xh = (half_t*)take((size_t)NN * 64 * 2);
  half_t* hnode = (half_t*)take((size_t)NN * 64 * 2);
  half_t* wn1h = (half_t*)take(64 * 64 * 2);
  half_t* wn2h = (half_t*)take(64 * 64 * 2);
  half_t* we1h = (half_t*)take(32 * 64 * 2);
  half_t* we2h = (half_t*)take(64 * 64 * 2);
  half_t* wm1h = (half_t*)take(128 * 64 * 2);
  half_t* wm2h = (half_t*)take(64 * 64 * 2);
  half_t* wu1h = (half_t*)take(64 * 64 * 2);
  half_t* wu2h = (half_t*)take(64 * 64 * 2);
  float* sums = (float*)take((size_t)(GG * 64 + GG) * 4);
  float* cnts = sums + (size_t)GG * 64;

  // f32 -> f16 staging (activations + weights); biases stay f32
  k_cvt<<<(NN * 64 + 255) / 256, 256, 0, stream>>>(node_features, xh, NN * 64);
  k_cvt<<<16, 256, 0, stream>>>(nw1, wn1h, 4096);
  k_cvt<<<16, 256, 0, stream>>>(nw2, wn2h, 4096);
  k_prep_we1<<<8, 256, 0, stream>>>(ew1, eb1, we1h);
  k_cvt<<<16, 256, 0, stream>>>(ew2, we2h, 4096);
  k_cvt<<<32, 256, 0, stream>>>(mw1, wm1h, 8192);
  k_cvt<<<16, 256, 0, stream>>>(mw2, wm2h, 4096);
  k_cvt<<<16, 256, 0, stream>>>(uw1, wu1h, 4096);
  k_cvt<<<16, 256, 0, stream>>>(uw2, wu2h, 4096);
  k_zero<<<(GG * 64 + GG + 255) / 256, 256, 0, stream>>>(sums, GG * 64 + GG);

  k_node_mlp<<<256, 256, 0, stream>>>(xh, wn1h, nb1, wn2h, nb2, hnode);
  k_edge<<<2048, 128, 0, stream>>>(node_pos, grid_pos, orient, eidx, eidx + EE,
                                   hnode, we1h, we2h, eb2, wm1h, mb1, wm2h,
                                   mb2, sums, cnts);
  k_finalize<<<128, 256, 0, stream>>>(sums, cnts, wu1h, ub1, wu2h, ub2,
                                      (float*)d_out);
}
